// PNA_90778428768716
// MI455X (gfx1250) — matrix operand order repaired and checked by reading, not compiled, not run on hardware
//
#include <hip/hip_runtime.h>
#include <hip/hip_bf16.h>
#include <math.h>

typedef __attribute__((ext_vector_type(16))) __bf16 v16bf;
typedef __attribute__((ext_vector_type(8)))  float  v8f;
typedef __attribute__((ext_vector_type(4)))  unsigned int u32x4;

union ABu { v16bf v; u32x4 q[2]; };

// A operand (16x32 bf16, M x K), row-major source with per-lane row base:
// lanes 0-15 hold K{0..7,16..23} (bytes +0,+32); lanes 16-31 hold K{8..15,24..31} (+16,+48)
__device__ __forceinline__ v16bf load_a16x32(const char* rowBase, int lane) {
  const char* q = rowBase + ((lane >> 4) << 4);
  ABu t;
  t.q[0] = *(const u32x4*)(q);
  t.q[1] = *(const u32x4*)(q + 32);
  return t.v;
}

// B operand (32x16 bf16, K x N) supplied as B^T row-major [16 cols][K...]:
// lane holds column (lane&15), K = 8*(lane>>4)+{0..7} and 16+8*(lane>>4)+{0..7} -> two b128 loads
__device__ __forceinline__ v16bf load_b32x16(const __bf16* btRow0, long strideElems, int lane) {
  const char* q = (const char*)(btRow0 + (long)(lane & 15) * strideElems) + ((lane >> 4) << 4);
  ABu t;
  t.q[0] = *(const u32x4*)(q);
  t.q[1] = *(const u32x4*)(q + 32);
  return t.v;
}

__device__ __forceinline__ v8f wmma_bf16(v16bf a, v16bf b, v8f c) {
  return __builtin_amdgcn_wmma_f32_16x16x32_bf16(false, a, false, b, (short)0, c, false, false);
}

// ---------------- prep: weight transpose->bf16, x->bf16, zero counters ----------------
__global__ void __launch_bounds__(256) pna_prep(
    const float* __restrict__ x, const float* __restrict__ pre_w,
    const float* __restrict__ post_w, const float* __restrict__ lin_w,
    __bf16* __restrict__ xbf, __bf16* __restrict__ pwt, __bf16* __restrict__ qwt,
    __bf16* __restrict__ lwt, int* __restrict__ degi, int* __restrict__ cursor,
    float* __restrict__ dls, int N)
{
  long i = (long)blockIdx.x * 256 + threadIdx.x;
  const long W0 = 64L * 128;            // pre_wt
  const long W1 = W0 + 64L * 832;       // post_wt
  const long W2 = W1 + 64L * 64;        // lin_wt  (= 65536)
  const long XR = W2 + (long)N * 64;    // xbf
  const long IR = XR + N;               // int counters
  if (i < W0) {
    long n = i / 128, k = i % 128;
    pwt[i] = (__bf16)pre_w[k * 64 + n];
  } else if (i < W1) {
    long j = i - W0; long n = j / 832, k = j % 832;
    qwt[j] = (__bf16)post_w[k * 64 + n];
  } else if (i < W2) {
    long j = i - W1; long n = j / 64, k = j % 64;
    lwt[j] = (__bf16)lin_w[k * 64 + n];
  } else if (i < XR) {
    long j = i - W2;
    xbf[j] = (__bf16)x[j];
  } else if (i < IR) {
    long j = i - XR;
    degi[j] = 0; cursor[j] = 0;
    if (j == 0) dls[0] = 0.f;
  }
}

// ---------------- degree histogram (int atomics) ----------------
__global__ void __launch_bounds__(256) pna_count(const int* __restrict__ ei,
                                                 int* __restrict__ degi, int E)
{
  int e = blockIdx.x * 256 + threadIdx.x;
  if (e < E) atomicAdd(&degi[ei[E + e]], 1);
}

// ---------------- exclusive prefix scan, single workgroup ----------------
__global__ void __launch_bounds__(256) pna_scan(const int* __restrict__ degi,
                                                int* __restrict__ rowptr, int N)
{
  __shared__ int sdata[256];
  __shared__ int carry;
  if (threadIdx.x == 0) carry = 0;
  __syncthreads();
  int nchunks = (N + 255) / 256;
  for (int c = 0; c < nchunks; ++c) {
    int i = c * 256 + threadIdx.x;
    int v = (i < N) ? degi[i] : 0;
    sdata[threadIdx.x] = v;
    __syncthreads();
    #pragma unroll
    for (int off = 1; off < 256; off <<= 1) {
      int t = (threadIdx.x >= off) ? sdata[threadIdx.x - off] : 0;
      __syncthreads();
      sdata[threadIdx.x] += t;
      __syncthreads();
    }
    int inc = sdata[threadIdx.x];           // inclusive scan of chunk
    int base = carry;
    if (i < N) rowptr[i] = base + inc - v;  // exclusive
    __syncthreads();
    if (threadIdx.x == 255) carry = base + inc;
    __syncthreads();
  }
  if (threadIdx.x == 0) rowptr[N] = carry;
}

// ---------------- CSR slot fill ----------------
__global__ void __launch_bounds__(256) pna_fill(const int* __restrict__ ei,
                                                const int* __restrict__ rowptr,
                                                int* __restrict__ cursor,
                                                int* __restrict__ eperm, int E)
{
  int e = blockIdx.x * 256 + threadIdx.x;
  if (e < E) {
    int d = ei[E + e];
    int slot = rowptr[d] + atomicAdd(&cursor[d], 1);
    eperm[slot] = e;
  }
}

// ---------------- edge message WMMA, permuted order -> contiguous mbuf ----------------
__global__ void __launch_bounds__(256) pna_edge_msg(
    const __bf16* __restrict__ xbf, const int* __restrict__ ei,
    const int* __restrict__ eperm, const __bf16* __restrict__ pwt,
    const float* __restrict__ pre_b, __bf16* __restrict__ mbuf, int E)
{
  int lane = threadIdx.x & 31;
  int wave = (blockIdx.x * blockDim.x + threadIdx.x) >> 5;
  int ntiles = (E + 15) >> 4;
  if (wave >= ntiles) return;          // wave-uniform: EXEC all-ones for WMMA
  int e0 = wave << 4;
  int el = e0 + (lane & 15); if (el >= E) el = E - 1;
  int ep   = eperm[el];                 // CSR-permuted edge id
  int srcN = ei[ep];                    // edge_index[0] = source/neighbor
  int dstN = ei[E + ep];                // edge_index[1] = target

  // h = [x_dst (64) | x_src (64)]  ->  m = h @ pre_w,  16 edges x 64 feats per wave
  v8f acc[4] = {};
  #pragma unroll
  for (int kc = 0; kc < 4; ++kc) {
    int node = (kc < 2) ? dstN : srcN;
    const char* arow = (const char*)(xbf + (long)node * 64 + (long)(kc & 1) * 32);
    v16bf a = load_a16x32(arow, lane);
    #pragma unroll
    for (int nt = 0; nt < 4; ++nt) {
      v16bf b = load_b32x16(pwt + (long)(nt * 16) * 128 + kc * 32, 128, lane);
      acc[nt] = wmma_bf16(a, b, acc[nt]);
    }
  }

  int half = lane >> 4;
  int nlo  = lane & 15;
  #pragma unroll
  for (int nt = 0; nt < 4; ++nt) {
    int f = nlo + nt * 16;
    float bias = pre_b[f];
    #pragma unroll
    for (int r = 0; r < 8; ++r) {
      int m = r + 8 * half;             // C/D row m = r + 8*(lane>>4)
      int e = e0 + m;
      if (e < E) mbuf[(long)e * 64 + f] = (__bf16)(acc[nt][r] + bias);
    }
  }
}

// ---------------- mean(log(deg+1)) reduction ----------------
__global__ void __launch_bounds__(256) pna_deglog(const int* __restrict__ rowptr,
                                                  float* __restrict__ dls, int N)
{
  int i = blockIdx.x * 256 + threadIdx.x;
  if (i < N) {
    float d = (float)(rowptr[i + 1] - rowptr[i]);
    __hip_atomic_fetch_add(dls, logf(d + 1.0f), __ATOMIC_RELAXED, __HIP_MEMORY_SCOPE_AGENT);
  }
}

// ---------------- per-node gather-reduce + finalize -> bf16 rows [N][832] ----------------
__global__ void __launch_bounds__(64) pna_gather(
    const float* __restrict__ x, const int* __restrict__ rowptr,
    const float* __restrict__ dls, const __bf16* __restrict__ mbuf,
    __bf16* __restrict__ hbuf, int N)
{
  int n = blockIdx.x;
  int f = threadIdx.x;
  if (n >= N) return;
  int r0 = rowptr[n], r1 = rowptr[n + 1];
  float s1 = 0.f, s2 = 0.f;
  float vmn = 3.402823466e38f, vmx = -3.402823466e38f;
  for (int j = r0; j < r1; ++j) {        // contiguous 128B rows, coalesced across 64 threads
    float v = (float)mbuf[(long)j * 64 + f];
    s1 += v; s2 += v * v;
    vmn = fminf(vmn, v); vmx = fmaxf(vmx, v);
  }
  float d = (float)(r1 - r0);
  float dc = fmaxf(d, 1.0f);
  float avg = dls[0] / (float)N;
  float dlog = logf(dc + 1.0f);
  float amp = dlog / avg;
  float att = avg / dlog;
  float mean = s1 / dc;
  float var = s2 / dc - mean * mean;
  if (var < 0.f) var = 0.f;
  float sd = sqrtf(var + 1e-5f);
  bool has = d > 0.f;
  if (!has) { vmn = 0.f; vmx = 0.f; }
  __bf16* row = hbuf + (long)n * 832;
  row[f]        = (__bf16)x[(long)n * 64 + f];
  row[64  + f]  = (__bf16)mean;
  row[128 + f]  = (__bf16)vmn;
  row[192 + f]  = (__bf16)vmx;
  row[256 + f]  = (__bf16)sd;
  row[320 + f]  = (__bf16)(mean * amp);
  row[384 + f]  = (__bf16)(vmn  * amp);
  row[448 + f]  = (__bf16)(vmx  * amp);
  row[512 + f]  = (__bf16)(sd   * amp);
  row[576 + f]  = (__bf16)(mean * att);
  row[640 + f]  = (__bf16)(vmn  * att);
  row[704 + f]  = (__bf16)(vmx  * att);
  row[768 + f]  = (__bf16)(sd   * att);
}

// ---------------- post MLP (832->64) + lin (64->64), fused, WMMA ----------------
__global__ void __launch_bounds__(256) pna_post_lin(
    const __bf16* __restrict__ hbuf, const __bf16* __restrict__ qwt,
    const float* __restrict__ post_b, const __bf16* __restrict__ lwt,
    const float* __restrict__ lin_b, float* __restrict__ out, int N)
{
  __shared__ __align__(16) __bf16 stage[8][16 * 64];   // per-wave 16x64 bf16 tile (16 KB)
  int lane = threadIdx.x & 31;
  int wib  = threadIdx.x >> 5;
  int ntiles = (N + 15) >> 4;
  int tile = blockIdx.x * 8 + wib;
  bool valid = tile < ntiles;
  int t = valid ? tile : 0;
  int row = t * 16 + (lane & 15); if (row >= N) row = N - 1;
  const char* arow = (const char*)(hbuf + (long)row * 832);

  v8f acc[4] = {};
  for (int kc = 0; kc < 26; ++kc) {                    // K = 832 = 26 * 32
    v16bf a = load_a16x32(arow + (long)kc * 64, lane);
    #pragma unroll
    for (int nt = 0; nt < 4; ++nt) {
      v16bf b = load_b32x16(qwt + (long)(nt * 16) * 832 + kc * 32, 832, lane);
      acc[nt] = wmma_bf16(a, b, acc[nt]);
    }
  }

  int half = lane >> 4;
  int nlo  = lane & 15;
  __bf16* st = stage[wib];
  #pragma unroll
  for (int nt = 0; nt < 4; ++nt) {
    float bias = post_b[nlo + nt * 16];
    #pragma unroll
    for (int r = 0; r < 8; ++r) {
      int m = r + 8 * half;
      st[m * 64 + nlo + nt * 16] = (__bf16)(acc[nt][r] + bias);
    }
  }
  __syncthreads();

  v8f acc2[4] = {};
  #pragma unroll
  for (int kc = 0; kc < 2; ++kc) {                     // K = 64 = 2 * 32
    v16bf a = load_a16x32((const char*)st + (long)(lane & 15) * 128 + kc * 64, lane);
    #pragma unroll
    for (int nt = 0; nt < 4; ++nt) {
      v16bf b = load_b32x16(lwt + (long)(nt * 16) * 64 + kc * 32, 64, lane);
      acc2[nt] = wmma_bf16(a, b, acc2[nt]);
    }
  }

  #pragma unroll
  for (int nt = 0; nt < 4; ++nt) {
    float bias = lin_b[nlo + nt * 16];
    #pragma unroll
    for (int r = 0; r < 8; ++r) {
      int m = r + 8 * half;
      int node = t * 16 + m;
      if (valid && node < N)
        out[(long)node * 64 + nlo + nt * 16] = acc2[nt][r] + bias;
    }
  }
}

extern "C" void kernel_launch(void* const* d_in, const int* in_sizes, int n_in,
                              void* d_out, int out_size, void* d_ws, size_t ws_size,
                              hipStream_t stream) {
  const float* x      = (const float*)d_in[0];
  const int*   ei     = (const int*)d_in[1];       // int32 (JAX x64 off); [2, E] row-major
  const float* pre_w  = (const float*)d_in[2];
  const float* pre_b  = (const float*)d_in[3];
  const float* post_w = (const float*)d_in[4];
  const float* post_b = (const float*)d_in[5];
  const float* lin_w  = (const float*)d_in[6];
  const float* lin_b  = (const float*)d_in[7];
  float* out = (float*)d_out;

  const int N = in_sizes[0] / 64;
  const int E = in_sizes[1] / 2;

  char* w = (char*)d_ws;
  auto take = [&](size_t bytes) { char* p = w; w += (bytes + 255) & ~(size_t)255; return p; };
  __bf16* xbf    = (__bf16*)take((size_t)N * 64 * 2);
  __bf16* pwt    = (__bf16*)take(64 * 128 * 2);
  __bf16* qwt    = (__bf16*)take(64 * 832 * 2);
  __bf16* lwt    = (__bf16*)take(64 * 64 * 2);
  int*    degi   = (int*)   take((size_t)N * 4);
  int*    cursor = (int*)   take((size_t)N * 4);
  int*    rowptr = (int*)   take((size_t)(N + 1) * 4);
  float*  dls    = (float*) take(256);
  int*    eperm  = (int*)   take((size_t)E * 4);
  __bf16* mbuf   = (__bf16*)take((size_t)E * 64 * 2);   // messages, CSR order (L2-resident)
  __bf16* hbuf   = (__bf16*)take((size_t)N * 832 * 2);

  // 1) prep: transpose weights to bf16, x->bf16, zero counters
  long prepTot = 65536L + (long)N * 64 + N;
  pna_prep<<<(int)((prepTot + 255) / 256), 256, 0, stream>>>(
      x, pre_w, post_w, lin_w, xbf, pwt, qwt, lwt, degi, cursor, dls, N);
  // 2) degree histogram
  pna_count<<<(E + 255) / 256, 256, 0, stream>>>(ei, degi, E);
  // 3) exclusive scan -> rowptr
  pna_scan<<<1, 256, 0, stream>>>(degi, rowptr, N);
  // 4) CSR fill -> eperm
  pna_fill<<<(E + 255) / 256, 256, 0, stream>>>(ei, rowptr, cursor, eperm, E);
  // 5) edge messages (WMMA) in CSR order -> contiguous mbuf (no float atomics)
  int etiles = (E + 15) / 16;
  int ethreads = etiles * 32;
  pna_edge_msg<<<(ethreads + 255) / 256, 256, 0, stream>>>(xbf, ei, eperm, pwt, pre_b, mbuf, E);
  // 6) sum of log(deg+1)
  pna_deglog<<<(N + 255) / 256, 256, 0, stream>>>(rowptr, dls, N);
  // 7) gather-reduce per node + finalize -> bf16 [N][832]
  pna_gather<<<N, 64, 0, stream>>>(x, rowptr, dls, mbuf, hbuf, N);
  // 8) post MLP + lin (WMMA), fused
  int ntiles = (N + 15) / 16;
  pna_post_lin<<<(ntiles + 7) / 8, 256, 0, stream>>>(hbuf, qwt, post_b, lwt, lin_b, out, N);
}
